// SimpleStateSpaceModel_54494545051921
// MI455X (gfx1250) — matrix operand order repaired and checked by reading, not compiled, not run on hardware
//
#include <hip/hip_runtime.h>
#include <hip/hip_bf16.h>
#include <math.h>

// Problem constants (from reference): B=4, S=4096, D=1024
#define BBATCH 4
#define SLEN   4096
#define DDIM   1024
#define BSROWS (BBATCH * SLEN)      // 16384 rows
#define NCHUNK 64                   // scan chunks per sequence
#define CLEN   (SLEN / NCHUNK)      // 64 steps per chunk

typedef __attribute__((ext_vector_type(16))) __bf16 v16bf;
typedef __attribute__((ext_vector_type(8)))  __bf16 v8bf;
typedef __attribute__((ext_vector_type(8)))  float  v8f;

__device__ __forceinline__ v8f wmma_bf16(v16bf a, v16bf b, v8f c) {
  // 8 args: (neg_a, A, neg_b, B, c_mod, C, reuse_a, reuse_b)
  return __builtin_amdgcn_wmma_f32_16x16x32_bf16(false, a, false, b, (short)0, c,
                                                 false, false);
}

union PackA { v16bf v; v8bf h[2]; };

// A-matrix 16x32 bf16 per-lane layout (contiguous K, matching B):
// lanes 0-15 : M=lane,    elems 0..15 -> K 0..15
// lanes 16-31: M=lane-16, elems 0..15 -> K 16..31
// => per lane: 16 contiguous elements at kbase,
//    kbase = kk + 16*hi
__device__ __forceinline__ v16bf load_a_tile(const __bf16* p) {
  PackA u;
  u.h[0] = *(const v8bf*)(p);
  u.h[1] = *(const v8bf*)(p + 8);
  return u.v;
}

// ---------------------------------------------------------------------------
// f32 -> bf16 conversion (grid-stride)
// ---------------------------------------------------------------------------
__global__ void cvt_f32_bf16(const float* __restrict__ src,
                             __bf16* __restrict__ dst, int n) {
  int i = blockIdx.x * blockDim.x + threadIdx.x;
  int stride = gridDim.x * blockDim.x;
  for (; i < n; i += stride) dst[i] = (__bf16)src[i];
}

// ---------------------------------------------------------------------------
// Phase 1: fused dt/B projections + gating epilogue.
// Each wave computes a 32(M) x 32(N) tile of BOTH z_dt = X @ dtW^T and
// z_B = X @ Bw^T (A operand shared): 8 x v8f accumulators (64 VGPRs), no
// spills. Then applies
//   dt   = softplus(z_dt + dt_b)
//   gate = exp(dt * (-exp(A_log)))
//   inp  = dt * z_B * x
// writing gate/inp (f32) for the scan.
// ---------------------------------------------------------------------------
__global__ void __launch_bounds__(256)
__attribute__((amdgpu_waves_per_eu(1, 8)))
proj_kernel(const __bf16* __restrict__ Xb,
            const __bf16* __restrict__ Wdt,
            const __bf16* __restrict__ Wbw,
            const float*  __restrict__ dt_b,
            const float*  __restrict__ A_log,
            const float*  __restrict__ Xf,
            float* __restrict__ gate,
            float* __restrict__ inp) {
  const int lane = threadIdx.x & 31;
  const int wave = blockIdx.x * (blockDim.x >> 5) + (threadIdx.x >> 5);
  const int nPairs = DDIM / 32;                 // 32
  const int mt2 = wave / nPairs;                // 0..511 (32-row block)
  const int np  = wave % nPairs;
  const int m0  = mt2 * 32;
  const int n0  = np * 32;

  const int lm = lane & 15;
  const int hi = lane >> 4;

  v8f accDt[2][2], accBw[2][2];
#pragma unroll
  for (int mi = 0; mi < 2; ++mi)
#pragma unroll
    for (int j = 0; j < 2; ++j) { accDt[mi][j] = {}; accBw[mi][j] = {}; }

  const __bf16* arow0 = Xb + (size_t)(m0 + lm) * DDIM;
  const __bf16* arow1 = Xb + (size_t)(m0 + 16 + lm) * DDIM;

  for (int kk = 0; kk < DDIM; kk += 32) {
    const int ka = kk + hi * 16;
    __builtin_prefetch(arow0 + ka + 32, 0, 3);  // speculative; drops if OOB
    __builtin_prefetch(arow1 + ka + 32, 0, 3);
    v16bf a0 = load_a_tile(arow0 + ka);
    v16bf a1 = load_a_tile(arow1 + ka);
    const int kb = kk + hi * 16;  // B layout: lane = col N, 16 contiguous K
#pragma unroll
    for (int j = 0; j < 2; ++j) {
      const int n = n0 + j * 16 + lm;
      v16bf bd = *(const v16bf*)(Wdt + (size_t)n * DDIM + kb);
      accDt[0][j] = wmma_bf16(a0, bd, accDt[0][j]);
      accDt[1][j] = wmma_bf16(a1, bd, accDt[1][j]);
      v16bf bb = *(const v16bf*)(Wbw + (size_t)n * DDIM + kb);
      accBw[0][j] = wmma_bf16(a0, bb, accBw[0][j]);
      accBw[1][j] = wmma_bf16(a1, bb, accBw[1][j]);
    }
  }

  // Epilogue. C/D layout: lane -> N = lm, vgpr r -> M = r + 8*hi.
#pragma unroll
  for (int j = 0; j < 2; ++j) {
    const int e   = n0 + j * 16 + lm;
    const float be = dt_b[e];
    const float Ae = -__expf(A_log[e]);
#pragma unroll
    for (int mi = 0; mi < 2; ++mi) {
#pragma unroll
      for (int r = 0; r < 8; ++r) {
        const int m = m0 + mi * 16 + r + hi * 8;
        const size_t o = (size_t)m * DDIM + e;
        float zdt = accDt[mi][j][r] + be;
        float dt  = (zdt > 20.f) ? zdt : log1pf(__expf(zdt));
        gate[o] = __expf(dt * Ae);
        inp[o]  = dt * accBw[mi][j][r] * Xf[o];
      }
    }
  }
}

// ---------------------------------------------------------------------------
// Phase 2a: per-chunk affine reduction. For each (b, chunk, d):
//   (a, v) <- compose over s in chunk of (g_s, u_s):  a*=g;  v = g*v + u
// ---------------------------------------------------------------------------
__global__ void scan_local(const float* __restrict__ gate,
                           const float* __restrict__ inp,
                           float* __restrict__ Ach,
                           float* __restrict__ Bch) {
  const int tid   = threadIdx.x;
  const int dtile = blockIdx.x & 3;                 // D/256 = 4 tiles
  const int ck    = (blockIdx.x >> 2) & (NCHUNK - 1);
  const int b     = blockIdx.x >> 8;                // 256 blocks per batch
  const int d     = dtile * 256 + tid;

  size_t base = ((size_t)b * SLEN + (size_t)ck * CLEN) * DDIM + d;
  float a = 1.f, v = 0.f;
  for (int s = 0; s < CLEN; ++s) {
    const float g = gate[base];
    const float u = inp[base];
    a *= g;
    v = __builtin_fmaf(g, v, u);
    base += DDIM;
  }
  const size_t o = ((size_t)b * NCHUNK + ck) * DDIM + d;
  Ach[o] = a;
  Bch[o] = v;
}

// ---------------------------------------------------------------------------
// Phase 2b: sequential combine across chunks (initial state = h0).
// Stores the hidden state at entry of each chunk.
// ---------------------------------------------------------------------------
__global__ void scan_chunks(const float* __restrict__ Ach,
                            const float* __restrict__ Bch,
                            const float* __restrict__ h0,
                            float* __restrict__ Entry) {
  const int idx = blockIdx.x * 256 + threadIdx.x;   // 0..B*D-1
  const int b = idx >> 10;
  const int d = idx & (DDIM - 1);
  float h = h0[(size_t)b * DDIM + d];
  for (int c = 0; c < NCHUNK; ++c) {
    const size_t o = ((size_t)b * NCHUNK + c) * DDIM + d;
    Entry[o] = h;
    h = __builtin_fmaf(Ach[o], h, Bch[o]);
  }
}

// ---------------------------------------------------------------------------
// Phase 2c: replay each chunk from its entry state, emit h as bf16 for the
// output GEMM.
// ---------------------------------------------------------------------------
__global__ void scan_apply(const float* __restrict__ gate,
                           const float* __restrict__ inp,
                           const float* __restrict__ Entry,
                           __bf16* __restrict__ Hb) {
  const int tid   = threadIdx.x;
  const int dtile = blockIdx.x & 3;
  const int ck    = (blockIdx.x >> 2) & (NCHUNK - 1);
  const int b     = blockIdx.x >> 8;
  const int d     = dtile * 256 + tid;

  float h = Entry[((size_t)b * NCHUNK + ck) * DDIM + d];
  size_t base = ((size_t)b * SLEN + (size_t)ck * CLEN) * DDIM + d;
  for (int s = 0; s < CLEN; ++s) {
    h = __builtin_fmaf(gate[base], h, inp[base]);
    Hb[base] = (__bf16)h;
    base += DDIM;
  }
}

// ---------------------------------------------------------------------------
// Phase 3: out = H @ Cw^T + Dp * x.  32(M) x 64(N) per wave, single output
// matrix -> 8 x v8f accumulators (64 VGPRs).
// ---------------------------------------------------------------------------
__global__ void __launch_bounds__(256)
__attribute__((amdgpu_waves_per_eu(1, 8)))
out_gemm(const __bf16* __restrict__ Hb,
         const __bf16* __restrict__ Wc,
         const float*  __restrict__ Dp,
         const float*  __restrict__ Xf,
         float* __restrict__ Out) {
  const int lane = threadIdx.x & 31;
  const int wave = blockIdx.x * (blockDim.x >> 5) + (threadIdx.x >> 5);
  const int nQuads = DDIM / 64;                 // 16
  const int mt2 = wave / nQuads;
  const int nq  = wave % nQuads;
  const int m0  = mt2 * 32;
  const int n0  = nq * 64;

  const int lm = lane & 15;
  const int hi = lane >> 4;

  v8f acc[2][4];
#pragma unroll
  for (int mi = 0; mi < 2; ++mi)
#pragma unroll
    for (int j = 0; j < 4; ++j) acc[mi][j] = {};

  const __bf16* arow0 = Hb + (size_t)(m0 + lm) * DDIM;
  const __bf16* arow1 = Hb + (size_t)(m0 + 16 + lm) * DDIM;

  for (int kk = 0; kk < DDIM; kk += 32) {
    const int ka = kk + hi * 16;
    __builtin_prefetch(arow0 + ka + 32, 0, 3);
    __builtin_prefetch(arow1 + ka + 32, 0, 3);
    v16bf a0 = load_a_tile(arow0 + ka);
    v16bf a1 = load_a_tile(arow1 + ka);
    const int kb = kk + hi * 16;
#pragma unroll
    for (int j = 0; j < 4; ++j) {
      const int n = n0 + j * 16 + lm;
      v16bf bm = *(const v16bf*)(Wc + (size_t)n * DDIM + kb);
      acc[0][j] = wmma_bf16(a0, bm, acc[0][j]);
      acc[1][j] = wmma_bf16(a1, bm, acc[1][j]);
    }
  }

#pragma unroll
  for (int j = 0; j < 4; ++j) {
    const int n = n0 + j * 16 + lm;
    const float dpn = Dp[n];
#pragma unroll
    for (int mi = 0; mi < 2; ++mi) {
#pragma unroll
      for (int r = 0; r < 8; ++r) {
        const int m = m0 + mi * 16 + r + hi * 8;
        const size_t o = (size_t)m * DDIM + n;
        Out[o] = acc[mi][j][r] + dpn * Xf[o];
      }
    }
  }
}

// ---------------------------------------------------------------------------
// Host launch
// ---------------------------------------------------------------------------
extern "C" void kernel_launch(void* const* d_in, const int* in_sizes, int n_in,
                              void* d_out, int out_size, void* d_ws,
                              size_t ws_size, hipStream_t stream) {
  (void)in_sizes; (void)n_in; (void)out_size; (void)ws_size;

  const float* x     = (const float*)d_in[0];  // [B,S,D]
  const float* h0    = (const float*)d_in[1];  // [B,D]
  const float* dt_w  = (const float*)d_in[2];  // [D,D]
  const float* dt_b  = (const float*)d_in[3];  // [D]
  const float* A_log = (const float*)d_in[4];  // [D]
  const float* B_w   = (const float*)d_in[5];  // [D,D]
  const float* C_w   = (const float*)d_in[6];  // [D,D]
  const float* Dp    = (const float*)d_in[7];  // [D]
  float* out = (float*)d_out;

  // Workspace carve-out (all 256B aligned by construction).
  char* w = (char*)d_ws;
  __bf16* xb   = (__bf16*)w;  w += (size_t)BSROWS * DDIM * 2;   // 32 MB
  __bf16* wdt  = (__bf16*)w;  w += (size_t)DDIM * DDIM * 2;     //  2 MB
  __bf16* wbw  = (__bf16*)w;  w += (size_t)DDIM * DDIM * 2;     //  2 MB
  __bf16* wcw  = (__bf16*)w;  w += (size_t)DDIM * DDIM * 2;     //  2 MB
  float*  gate = (float*)w;   w += (size_t)BSROWS * DDIM * 4;   // 64 MB
  float*  inp  = (float*)w;   w += (size_t)BSROWS * DDIM * 4;   // 64 MB
  float*  Ach  = (float*)w;   w += (size_t)BBATCH * NCHUNK * DDIM * 4;
  float*  Bch  = (float*)w;   w += (size_t)BBATCH * NCHUNK * DDIM * 4;
  float*  Ent  = (float*)w;   w += (size_t)BBATCH * NCHUNK * DDIM * 4;
  __bf16* hb   = (__bf16*)w;  w += (size_t)BSROWS * DDIM * 2;   // 32 MB

  // 0) bf16 casts
  cvt_f32_bf16<<<2048, 256, 0, stream>>>(x, xb, BSROWS * DDIM);
  cvt_f32_bf16<<<256, 256, 0, stream>>>(dt_w, wdt, DDIM * DDIM);
  cvt_f32_bf16<<<256, 256, 0, stream>>>(B_w, wbw, DDIM * DDIM);
  cvt_f32_bf16<<<256, 256, 0, stream>>>(C_w, wcw, DDIM * DDIM);

  // 1) fused dt/B projections + gate/input epilogue
  //    waves = (BS/32)*(D/32) = 16384 -> 2048 blocks x 8 waves
  proj_kernel<<<2048, 256, 0, stream>>>(xb, wdt, wbw, dt_b, A_log, x, gate, inp);

  // 2) chunked affine scan
  scan_local<<<BBATCH * NCHUNK * (DDIM / 256), 256, 0, stream>>>(gate, inp, Ach, Bch);
  scan_chunks<<<(BBATCH * DDIM) / 256, 256, 0, stream>>>(Ach, Bch, h0, Ent);
  scan_apply<<<BBATCH * NCHUNK * (DDIM / 256), 256, 0, stream>>>(gate, inp, Ent, hb);

  // 3) output projection + skip term
  //    waves = (BS/32)*(D/64) = 8192 -> 1024 blocks x 8 waves
  out_gemm<<<1024, 256, 0, stream>>>(hb, wcw, Dp, x, out);
}
